// PCADetector_29008209117383
// MI455X (gfx1250) — matrix operand order repaired and checked by reading, not compiled, not run on hardware
//
#include <hip/hip_runtime.h>
#include <hip/hip_bf16.h>

typedef __attribute__((ext_vector_type(16))) _Float16 v16h;
typedef __attribute__((ext_vector_type(8)))  float    v8f;

#define NBINS 256
#define BATCH 64
#define HW (512 * 512)

// ---------------------------------------------------------------------------
// Kernel 0: zero the 64x256 histogram counters in workspace (poisoned 0xAA).
// ---------------------------------------------------------------------------
__global__ void zero_cnt_kernel(unsigned int* __restrict__ cnt) {
    cnt[blockIdx.x * 256 + threadIdx.x] = 0u;
}

// ---------------------------------------------------------------------------
// Kernel 1: per-image 256-bin histogram of channel 0 (torch.histc semantics
// over [0,1]). grid = (16, 64): block (i, b) handles floats
// [i*16384, (i+1)*16384) of image b's channel-0 plane (contiguous -> float4).
// LDS-private histogram, then one global atomic per bin per block.
// ---------------------------------------------------------------------------
__global__ void __launch_bounds__(256)
hist_kernel(const float* __restrict__ x, unsigned int* __restrict__ cnt) {
    __shared__ unsigned int lh[NBINS];
    const int t = threadIdx.x;
    lh[t] = 0u;
    __syncthreads();

    const int b = blockIdx.y;
    const float4* base =
        (const float4*)(x + (size_t)b * 3 * HW) + (size_t)blockIdx.x * 4096 + t;
    #pragma unroll
    for (int i = 0; i < 16; ++i) {
        float4 v = base[i * 256];
        float vv[4] = {v.x, v.y, v.z, v.w};
        #pragma unroll
        for (int j = 0; j < 4; ++j) {
            float f = vv[j];
            if (f >= 0.0f && f <= 1.0f) {
                int bin = (int)(f * 256.0f);          // floor for f >= 0
                bin = bin > 255 ? 255 : bin;
                atomicAdd(&lh[bin], 1u);
            }
        }
    }
    __syncthreads();
    atomicAdd(&cnt[b * NBINS + t], lh[t]);
}

// ---------------------------------------------------------------------------
// WMMA fragment helpers (wave32; A and B share contiguous k = 16*(lane/16) + elem, not the ISA table order)
// ---------------------------------------------------------------------------
// A (16x32 f16, MxK), row-major source with leading dim lda:
//   lanes 0-15: M=lane, elems 0-15 -> K=kbase+0..15
//   lanes 16-31: M=lane-16, elems 0-15 -> K=kbase+16..31
__device__ __forceinline__ v16h load_a_frag(const _Float16* A, int lda,
                                            int mbase, int kbase, int lane) {
    const int l  = lane & 15;
    const int hi = (lane >> 4) & 1;
    const _Float16* row = A + (mbase + l) * lda + kbase + 16 * hi;
    v16h a;
    #pragma unroll
    for (int i = 0; i < 8; ++i) a[i] = row[i];
    #pragma unroll
    for (int i = 0; i < 8; ++i) a[8 + i] = row[8 + i];
    return a;
}

// B (32x16 f16, KxN) taken as W^T where W is row-major (N_out x K_in), ldw:
//   lane -> N = nbase + (lane&15); elem i -> K = kbase + i + 16*(lane>=16)
__device__ __forceinline__ v16h load_bt_frag(const float* W, int ldw,
                                             int nbase, int kbase, int lane) {
    const int n  = lane & 15;
    const int hi = (lane >> 4) & 1;
    const float* row = W + (nbase + n) * ldw + kbase + 16 * hi;
    v16h b;
    #pragma unroll
    for (int i = 0; i < 16; ++i) b[i] = (_Float16)row[i];
    return b;
}

// C/D (16x16 f32): lane -> N = nbase + (lane&15); VGPR r -> M = r + 8*(lane>=16)
__device__ __forceinline__ void store_c(float* Y, int ldy, int mbase, int nbase,
                                        v8f c, const float* bias, int lane) {
    const int n  = lane & 15;
    const int hi = (lane >> 4) & 1;
    const float bv = bias[nbase + n];
    #pragma unroll
    for (int r = 0; r < 8; ++r)
        Y[(mbase + r + 8 * hi) * ldy + nbase + n] = c[r] + bv;
}

// ---------------------------------------------------------------------------
// Kernel 2: normalize histograms + full MLP in one workgroup (8 wave32s).
// LDS reuse: [0,32K)=hist f16 then A2 f16; [32K,64K)=Y1 f32 then Y2 f32.
// ---------------------------------------------------------------------------
__global__ void __launch_bounds__(256)
mlp_kernel(const unsigned int* __restrict__ cnt,
           const float* __restrict__ W1, const float* __restrict__ b1,
           const float* __restrict__ g1, const float* __restrict__ be1,
           const float* __restrict__ W2, const float* __restrict__ b2,
           const float* __restrict__ g2, const float* __restrict__ be2,
           const float* __restrict__ W3, const float* __restrict__ b3,
           float* __restrict__ out) {
    __shared__ __align__(16) unsigned char smem[65536];
    _Float16* histA = (_Float16*)smem;             // [64][256] f16
    float*    Y1    = (float*)(smem + 32768);      // [64][128] f32
    _Float16* A2    = (_Float16*)smem;             // [64][128] f16 (reuse)
    float*    Y2    = (float*)(smem + 32768);      // [64][64]  f32 (reuse)

    const int t    = threadIdx.x;
    const int lane = t & 31;
    const int wv   = t >> 5;

    // Phase 0: row-normalize counts -> f16 activations (hist / (sum + 1e-6))
    if (t < BATCH) {
        float s = 0.0f;
        for (int k = 0; k < NBINS; ++k) s += (float)cnt[t * NBINS + k];
        const float inv = 1.0f / (s + 1e-6f);
        for (int k = 0; k < NBINS; ++k)
            histA[t * NBINS + k] = (_Float16)((float)cnt[t * NBINS + k] * inv);
    }
    __syncthreads();

    // Layer 1: (64x256) @ W1^T(256x128). Wave w owns N-tile w; 4 M-tiles, K=8x32.
    {
        const int nbase = wv * 16;
        for (int m = 0; m < 4; ++m) {
            v8f c = {};
            #pragma unroll
            for (int k = 0; k < 8; ++k) {
                v16h a = load_a_frag(histA, 256, m * 16, k * 32, lane);
                v16h b = load_bt_frag(W1, 256, nbase, k * 32, lane);
                c = __builtin_amdgcn_wmma_f32_16x16x32_f16(
                        false, a, false, b, (short)0, c, false, false);
            }
            store_c(Y1, 128, m * 16, nbase, c, b1, lane);
        }
    }
    __syncthreads();

    // BN1 (biased batch stats) + ReLU -> A2 (f16)
    if (t < 128) {
        float s = 0.0f, s2 = 0.0f;
        for (int b = 0; b < BATCH; ++b) {
            float v = Y1[b * 128 + t]; s += v; s2 += v * v;
        }
        const float mu  = s * (1.0f / BATCH);
        const float var = s2 * (1.0f / BATCH) - mu * mu;
        const float sc  = rsqrtf(var + 1e-5f) * g1[t];
        const float sh  = be1[t];
        for (int b = 0; b < BATCH; ++b) {
            float v = (Y1[b * 128 + t] - mu) * sc + sh;
            A2[b * 128 + t] = (_Float16)fmaxf(v, 0.0f);
        }
    }
    __syncthreads();

    // Layer 2: (64x128) @ W2^T(128x64). 16 tiles, 2 per wave; K=4x32.
    {
        #pragma unroll
        for (int tt = 0; tt < 2; ++tt) {
            const int tile = wv * 2 + tt;
            const int m = tile >> 2, n = tile & 3;
            v8f c = {};
            #pragma unroll
            for (int k = 0; k < 4; ++k) {
                v16h a = load_a_frag(A2, 128, m * 16, k * 32, lane);
                v16h b = load_bt_frag(W2, 128, n * 16, k * 32, lane);
                c = __builtin_amdgcn_wmma_f32_16x16x32_f16(
                        false, a, false, b, (short)0, c, false, false);
            }
            store_c(Y2, 64, m * 16, n * 16, c, b2, lane);
        }
    }
    __syncthreads();

    // BN2 + ReLU in place (each thread owns one column)
    if (t < BATCH) {
        float s = 0.0f, s2 = 0.0f;
        for (int b = 0; b < BATCH; ++b) {
            float v = Y2[b * 64 + t]; s += v; s2 += v * v;
        }
        const float mu  = s * (1.0f / BATCH);
        const float var = s2 * (1.0f / BATCH) - mu * mu;
        const float sc  = rsqrtf(var + 1e-5f) * g2[t];
        const float sh  = be2[t];
        for (int b = 0; b < BATCH; ++b) {
            float v = (Y2[b * 64 + t] - mu) * sc + sh;
            Y2[b * 64 + t] = fmaxf(v, 0.0f);
        }
    }
    __syncthreads();

    // Layer 3: (64x64) @ W3^T(64x1) + b3
    if (t < BATCH) {
        float s = b3[0];
        for (int k = 0; k < 64; ++k) s += Y2[t * 64 + k] * W3[k];
        out[t] = s;
    }
}

// ---------------------------------------------------------------------------
extern "C" void kernel_launch(void* const* d_in, const int* in_sizes, int n_in,
                              void* d_out, int out_size, void* d_ws, size_t ws_size,
                              hipStream_t stream) {
    const float* x   = (const float*)d_in[0];
    const float* W1  = (const float*)d_in[1];
    const float* b1  = (const float*)d_in[2];
    const float* g1  = (const float*)d_in[3];
    const float* be1 = (const float*)d_in[4];
    const float* W2  = (const float*)d_in[5];
    const float* b2  = (const float*)d_in[6];
    const float* g2  = (const float*)d_in[7];
    const float* be2 = (const float*)d_in[8];
    const float* W3  = (const float*)d_in[9];
    const float* b3  = (const float*)d_in[10];
    float* out = (float*)d_out;

    unsigned int* cnt = (unsigned int*)d_ws;   // 64*256 u32 = 64 KB

    zero_cnt_kernel<<<64, 256, 0, stream>>>(cnt);
    hist_kernel<<<dim3(16, 64), 256, 0, stream>>>(x, cnt);
    mlp_kernel<<<1, 256, 0, stream>>>(cnt, W1, b1, g1, be1,
                                      W2, b2, g2, be2, W3, b3, out);
}
